// MultiHeadAttention_2508260901282
// MI455X (gfx1250) — matrix operand order repaired and checked by reading, not compiled, not run on hardware
//
#include <hip/hip_runtime.h>
#include <hip/hip_bf16.h>

typedef __bf16 bf16_t;
typedef __attribute__((ext_vector_type(16))) __bf16 v16bf;
typedef __attribute__((ext_vector_type(8)))  __bf16 v8bf;
typedef __attribute__((ext_vector_type(8)))  float   v8f;

#define BATCH 4
#define SEQ   2048
#define CDIM  1024
#define NHEAD 16
#define HDIM  64
#define BT    (BATCH * SEQ)      // 8192
#define BH    (BATCH * NHEAD)    // 64

#define TILE_M 64
#define TILE_N 128
#define TILE_K 32
#define LDA 40   // 32 + 8 bf16 pad (keeps 16B alignment)
#define LDB 40

#if defined(__AMDGCN__) && __has_builtin(__builtin_amdgcn_tensor_load_to_lds)
#define HAVE_TDM 1
#else
#define HAVE_TDM 0
#endif

#if HAVE_TDM
typedef __attribute__((ext_vector_type(4))) unsigned int u32x4;
typedef __attribute__((ext_vector_type(8))) int          i32x8;
typedef __attribute__((ext_vector_type(4))) int          i32x4;

// Issue one TDM 2D tile load: global (row-major, stride_x elements of 2B) -> LDS
// (packed tile_x * tile_y). Descriptor bit layout per CDNA5 ISA 8.3/8.4.
// 6-arg builtin form (clang-23 / therock-10.0): (g0, g1, g2, g3, g_extra, cpol).
static __device__ inline void tdm_load_2d(const void* gptr, unsigned lds_off,
                                          unsigned tile_x, unsigned tile_y,
                                          unsigned tensor_x, unsigned tensor_y,
                                          unsigned stride_x) {
  const unsigned long long ga = (unsigned long long)(uintptr_t)gptr;
  u32x4 g0;
  g0[0] = 1u;                                   // count=1, user descriptor
  g0[1] = lds_off;                              // lds_addr
  g0[2] = (unsigned)ga;                         // global_addr[31:0]
  g0[3] = (unsigned)(ga >> 32) | (2u << 30);    // global_addr[57:32] | type=2
  i32x8 g1;
  g1[0] = (1 << 16);                            // wg_mask=0, data_size=1 (2 bytes)
  g1[1] = (int)((tensor_x & 0xFFFFu) << 16);    // tensor_dim0[15:0] @ [63:48]
  g1[2] = (int)((tensor_x >> 16) | ((tensor_y & 0xFFFFu) << 16));
  g1[3] = (int)((tensor_y >> 16) | (tile_x << 16));   // tile_dim0 @ [127:112]
  g1[4] = (int)tile_y;                          // tile_dim1; tile_dim2=0
  g1[5] = (int)stride_x;                        // tensor_dim0_stride[31:0]
  g1[6] = 0;                                    // stride hi / dim1_stride lo
  g1[7] = 0;
  i32x4 z4;
  z4[0] = 0; z4[1] = 0; z4[2] = 0; z4[3] = 0;
  i32x8 z8;
#pragma unroll
  for (int i = 0; i < 8; ++i) z8[i] = 0;
  __builtin_amdgcn_tensor_load_to_lds(g0, g1, z4, z4, z8, 0);
}
#endif

static __device__ inline v8f zero_v8f() {
  v8f z;
#pragma unroll
  for (int i = 0; i < 8; ++i) z[i] = 0.0f;
  return z;
}

static __device__ inline v16bf combine16(v8bf lo, v8bf hi) {
  v16bf r;
#pragma unroll
  for (int i = 0; i < 8; ++i) { r[i] = lo[i]; r[i + 8] = hi[i]; }
  return r;
}

// A-matrix fragment (16x32 bf16, MxK). Lanes 0-15: rows M=0..15, K {0..7,16..23};
// lanes 16-31: same rows, K {8..15,24..31}.
static __device__ inline v16bf load_a_frag(const bf16_t* __restrict__ base, int ld) {
  const int lane = threadIdx.x & 31;
  const int row  = lane & 15;
  const int ko   = (lane >> 4) << 3;
  const bf16_t* p = base + row * ld + ko;
  v8bf lo = *(const v8bf*)(p);
  v8bf hi = *(const v8bf*)(p + 16);
  return combine16(lo, hi);
}

// B-matrix fragment (32x16 bf16, KxN) from memory where B column n is a
// contiguous-K row. Lanes 0-15: N with K {0..7,16..23}; lanes 16-31: K {8..15,24..31}.
static __device__ inline v16bf load_b_frag(const bf16_t* __restrict__ base, int ld) {
  const int lane = threadIdx.x & 31;
  const int row  = lane & 15;
  const int ko   = (lane >> 4) << 3;
  const bf16_t* p = base + row * ld + ko;
  v8bf lo = *(const v8bf*)(p);
  v8bf hi = *(const v8bf*)(p + 16);
  return combine16(lo, hi);
}

static __device__ inline v8f wmma_bf16(v16bf a, v16bf b, v8f c) {
  return __builtin_amdgcn_wmma_f32_16x16x32_bf16(false, a, false, b, (short)0, c,
                                                 false, false);
}

static __device__ inline void cvt_store4(bf16_t* d, float4 f) {
  d[0] = (bf16_t)f.x; d[1] = (bf16_t)f.y; d[2] = (bf16_t)f.z; d[3] = (bf16_t)f.w;
}

// ---------------------------------------------------------------------------
// Kernel 1: projection GEMM  Y = X * W^T  (fp32 in, bf16 out, per-head layout)
// mode 0: dst[((b*H+h)*T + t)*64 + d]    (Q, K)
// mode 2: dst[((b*H+h)*64 + d)*T + t]    (V transposed for P*V fragments)
// ---------------------------------------------------------------------------
__global__ __launch_bounds__(256) void proj_gemm_kernel(
    const float* __restrict__ X, const float* __restrict__ W,
    bf16_t* __restrict__ dst, int mode) {
  __shared__ __align__(16) bf16_t lA[TILE_M * LDA];
  __shared__ __align__(16) bf16_t lB[TILE_N * LDB];

  const int tid  = threadIdx.x;
  const int lane = tid & 31;
  const int wave = tid >> 5;
  const int wm = wave & 1;
  const int wn = wave >> 1;
  const int mBase = blockIdx.x * TILE_M;
  const int nBase = blockIdx.y * TILE_N;
  const int srow = tid >> 3;        // 0..31
  const int scol = (tid & 7) * 4;   // 0..28

  v8f acc[2][2];
#pragma unroll
  for (int i = 0; i < 2; ++i)
#pragma unroll
    for (int j = 0; j < 2; ++j) acc[i][j] = zero_v8f();

  for (int k0 = 0; k0 < CDIM; k0 += TILE_K) {
    // Issue ALL tile loads first (6 x b128 in flight), then convert + ds_store.
    float4 fa[2], fb[4];
#pragma unroll
    for (int p = 0; p < 2; ++p)
      fa[p] = *(const float4*)(X + (size_t)(mBase + srow + p * 32) * CDIM + k0 + scol);
#pragma unroll
    for (int p = 0; p < 4; ++p)
      fb[p] = *(const float4*)(W + (size_t)(nBase + srow + p * 32) * CDIM + k0 + scol);
#pragma unroll
    for (int p = 0; p < 2; ++p)
      cvt_store4(&lA[(srow + p * 32) * LDA + scol], fa[p]);
#pragma unroll
    for (int p = 0; p < 4; ++p)
      cvt_store4(&lB[(srow + p * 32) * LDB + scol], fb[p]);
    // Prefetch next K-step tiles into cache while we compute on this one.
    if (k0 + TILE_K < CDIM) {
      __builtin_prefetch(X + (size_t)(mBase + srow) * CDIM + k0 + TILE_K + scol, 0, 1);
      __builtin_prefetch(W + (size_t)(nBase + srow) * CDIM + k0 + TILE_K + scol, 0, 1);
    }
    __syncthreads();

    v16bf aF[2], bF[2];
#pragma unroll
    for (int i = 0; i < 2; ++i) aF[i] = load_a_frag(&lA[(wm * 32 + i * 16) * LDA], LDA);
#pragma unroll
    for (int j = 0; j < 2; ++j) bF[j] = load_b_frag(&lB[(wn * 32 + j * 16) * LDB], LDB);
#pragma unroll
    for (int i = 0; i < 2; ++i)
#pragma unroll
      for (int j = 0; j < 2; ++j) acc[i][j] = wmma_bf16(aF[i], bF[j], acc[i][j]);
    __syncthreads();
  }

  const int half = lane >> 4, ln = lane & 15;
#pragma unroll
  for (int i = 0; i < 2; ++i)
#pragma unroll
    for (int j = 0; j < 2; ++j)
#pragma unroll
      for (int r = 0; r < 8; ++r) {
        const int m = mBase + wm * 32 + i * 16 + half * 8 + r;
        const int o = nBase + wn * 32 + j * 16 + ln;
        const int b = m >> 11, t = m & (SEQ - 1);
        const int h = o >> 6, d = o & 63;
        const size_t idx = (mode == 2)
            ? ((((size_t)b * NHEAD + h) * HDIM + d) * SEQ + t)
            : ((((size_t)b * NHEAD + h) * SEQ + t) * HDIM + d);
        dst[idx] = (bf16_t)acc[i][j][r];
      }
}

// ---------------------------------------------------------------------------
// Kernel 2: flash attention per (b,h). 8 waves x 16 query rows per block.
// K/V tiles staged once per kv step into LDS via TDM async DMA (wave 0 builds
// the D# descriptor, issues tensor_load_to_lds, s_wait_tensorcnt), shared by
// all 8 waves -> 8x less L2 traffic than direct per-wave loads.
// Q,K: [BH][T][64] bf16 ; VT: [BH][64][T] bf16 ; O: [B][T][C] bf16
// ---------------------------------------------------------------------------
__global__ __launch_bounds__(256) void attn_kernel(
    const bf16_t* __restrict__ Q, const bf16_t* __restrict__ K,
    const bf16_t* __restrict__ VT, bf16_t* __restrict__ O) {
  __shared__ __align__(16) bf16_t ldsK[64 * HDIM];    // [kv_local][d]   8KB
  __shared__ __align__(16) bf16_t ldsV[HDIM * 64];    // [d][kv_local]   8KB
  __shared__ __align__(16) bf16_t pS[8][16 * HDIM];   // per-wave P     16KB

  const int tid  = threadIdx.x;
  const int lane = tid & 31;
  const int wave = tid >> 5;
  const int half = lane >> 4, ln = lane & 15;
  const int bh = blockIdx.y;
  const int qBase = blockIdx.x * 128 + wave * 16;

  const bf16_t* Qb = Q + ((size_t)bh * SEQ + qBase) * HDIM;
  const bf16_t* Kb = K + (size_t)bh * SEQ * HDIM;
  const bf16_t* Vb = VT + (size_t)bh * HDIM * SEQ;

  v16bf qF[2];
  qF[0] = load_a_frag(Qb + 0, HDIM);
  qF[1] = load_a_frag(Qb + 32, HDIM);

  v8f oacc[4];
#pragma unroll
  for (int n = 0; n < 4; ++n) oacc[n] = zero_v8f();
  float mrow[8], lrow[8];
#pragma unroll
  for (int r = 0; r < 8; ++r) { mrow[r] = -1e30f; lrow[r] = 0.0f; }

  const float scale = 0.125f;  // 1/sqrt(64)

  for (int kv = 0; kv < SEQ; kv += 64) {
    __syncthreads();  // previous iteration done consuming ldsK/ldsV
#if HAVE_TDM
    if (wave == 0) {
      tdm_load_2d(Kb + (size_t)kv * HDIM,
                  (unsigned)(uintptr_t)(void*)&ldsK[0],
                  /*tile_x=*/HDIM, /*tile_y=*/64,
                  /*tensor_x=*/HDIM, /*tensor_y=*/SEQ, /*stride_x=*/HDIM);
      tdm_load_2d(Vb + kv,
                  (unsigned)(uintptr_t)(void*)&ldsV[0],
                  /*tile_x=*/64, /*tile_y=*/HDIM,
                  /*tensor_x=*/SEQ, /*tensor_y=*/HDIM, /*stride_x=*/SEQ);
      __builtin_amdgcn_s_wait_tensorcnt(0);
    }
#else
    {
      // Batch all 4 tile loads before any LDS store.
      const bf16_t* ks = Kb + (size_t)kv * HDIM;
      const int e = tid * 16;
      const int vr = tid >> 2;
      const int vc = (tid & 3) * 16;
      const bf16_t* vs = Vb + (size_t)vr * SEQ + kv + vc;
      v8bf k0v = *(const v8bf*)(ks + e);
      v8bf k1v = *(const v8bf*)(ks + e + 8);
      v8bf v0v = *(const v8bf*)(vs);
      v8bf v1v = *(const v8bf*)(vs + 8);
      *(v8bf*)&ldsK[e]            = k0v;
      *(v8bf*)&ldsK[e + 8]        = k1v;
      *(v8bf*)&ldsV[vr * 64 + vc]     = v0v;
      *(v8bf*)&ldsV[vr * 64 + vc + 8] = v1v;
    }
#endif
    __syncthreads();  // tiles visible to all waves

    // --- S = scale * Q * K^T  (16 x 64 tile per wave) ---
    v8f s[4];
#pragma unroll
    for (int nt = 0; nt < 4; ++nt) {
      const bf16_t* kp = &ldsK[(nt * 16) * HDIM];
      v8f a = wmma_bf16(qF[0], load_b_frag(kp + 0, HDIM), zero_v8f());
      a = wmma_bf16(qF[1], load_b_frag(kp + 32, HDIM), a);
      s[nt] = a * scale;
    }

    // --- online softmax: row max (row stats share the C-layout lane mapping) ---
    float tmax[8];
#pragma unroll
    for (int r = 0; r < 8; ++r) {
      float t0 = fmaxf(fmaxf(s[0][r], s[1][r]), fmaxf(s[2][r], s[3][r]));
      t0 = fmaxf(t0, __shfl_xor(t0, 1, 32));
      t0 = fmaxf(t0, __shfl_xor(t0, 2, 32));
      t0 = fmaxf(t0, __shfl_xor(t0, 4, 32));
      t0 = fmaxf(t0, __shfl_xor(t0, 8, 32));
      tmax[r] = t0;
    }
    float alpha[8];
#pragma unroll
    for (int r = 0; r < 8; ++r) {
      const float mnew = fmaxf(mrow[r], tmax[r]);
      alpha[r] = __expf(mrow[r] - mnew);
      mrow[r] = mnew;
    }

    // --- P = exp(S - m), row sums, stage P to LDS (C-layout -> A-layout) ---
    float rsum[8];
#pragma unroll
    for (int r = 0; r < 8; ++r) rsum[r] = 0.0f;
#pragma unroll
    for (int nt = 0; nt < 4; ++nt)
#pragma unroll
      for (int r = 0; r < 8; ++r) {
        const float p = __expf(s[nt][r] - mrow[r]);
        rsum[r] += p;
        pS[wave][(half * 8 + r) * HDIM + nt * 16 + ln] = (bf16_t)p;
      }
#pragma unroll
    for (int r = 0; r < 8; ++r) {
      float t0 = rsum[r];
      t0 += __shfl_xor(t0, 1, 32);
      t0 += __shfl_xor(t0, 2, 32);
      t0 += __shfl_xor(t0, 4, 32);
      t0 += __shfl_xor(t0, 8, 32);
      lrow[r] = lrow[r] * alpha[r] + t0;
    }
#pragma unroll
    for (int n = 0; n < 4; ++n)
#pragma unroll
      for (int r = 0; r < 8; ++r) oacc[n][r] *= alpha[r];

    __syncthreads();  // cross-lane P visibility

    // --- O += P * V  (A = P from LDS, B = V^T tile from LDS) ---
    v16bf pF0 = load_a_frag(&pS[wave][0], HDIM);
    v16bf pF1 = load_a_frag(&pS[wave][32], HDIM);
#pragma unroll
    for (int dn = 0; dn < 4; ++dn) {
      const bf16_t* vp = &ldsV[(dn * 16) * 64];
      oacc[dn] = wmma_bf16(pF0, load_b_frag(vp + 0, 64), oacc[dn]);
      oacc[dn] = wmma_bf16(pF1, load_b_frag(vp + 32, 64), oacc[dn]);
    }
  }

  // --- epilogue: divide by l, store [B][T][C] bf16 ---
  const int b = bh >> 4, h = bh & 15;
#pragma unroll
  for (int dn = 0; dn < 4; ++dn)
#pragma unroll
    for (int r = 0; r < 8; ++r) {
      const int t = qBase + half * 8 + r;
      const int d = dn * 16 + ln;
      O[((size_t)b * SEQ + t) * CDIM + h * HDIM + d] =
          (bf16_t)(oacc[dn][r] / lrow[r]);
    }
}

// ---------------------------------------------------------------------------
// Kernel 3: output projection  out = A * Wo^T + bo  (bf16 A, fp32 out)
// ---------------------------------------------------------------------------
__global__ __launch_bounds__(256) void out_gemm_kernel(
    const bf16_t* __restrict__ A, const float* __restrict__ W,
    const float* __restrict__ bias, float* __restrict__ out) {
  __shared__ __align__(16) bf16_t lA[TILE_M * LDA];
  __shared__ __align__(16) bf16_t lB[TILE_N * LDB];

  const int tid  = threadIdx.x;
  const int lane = tid & 31;
  const int wave = tid >> 5;
  const int wm = wave & 1;
  const int wn = wave >> 1;
  const int mBase = blockIdx.x * TILE_M;
  const int nBase = blockIdx.y * TILE_N;
  const int srow = tid >> 3;
  const int scol = (tid & 7) * 4;
  const int arow = tid >> 2;
  const int acol = (tid & 3) * 8;

  v8f acc[2][2];
#pragma unroll
  for (int i = 0; i < 2; ++i)
#pragma unroll
    for (int j = 0; j < 2; ++j) acc[i][j] = zero_v8f();

  for (int k0 = 0; k0 < CDIM; k0 += TILE_K) {
    // Issue all 5 tile loads first, then convert + store.
    v8bf av = *(const v8bf*)(A + (size_t)(mBase + arow) * CDIM + k0 + acol);
    float4 fb[4];
#pragma unroll
    for (int p = 0; p < 4; ++p)
      fb[p] = *(const float4*)(W + (size_t)(nBase + srow + p * 32) * CDIM + k0 + scol);
    *(v8bf*)(&lA[arow * LDA + acol]) = av;
#pragma unroll
    for (int p = 0; p < 4; ++p)
      cvt_store4(&lB[(srow + p * 32) * LDB + scol], fb[p]);
    if (k0 + TILE_K < CDIM) {
      __builtin_prefetch(A + (size_t)(mBase + arow) * CDIM + k0 + TILE_K + acol, 0, 1);
      __builtin_prefetch(W + (size_t)(nBase + srow) * CDIM + k0 + TILE_K + scol, 0, 1);
    }
    __syncthreads();

    v16bf aF[2], bF[2];
#pragma unroll
    for (int i = 0; i < 2; ++i) aF[i] = load_a_frag(&lA[(wm * 32 + i * 16) * LDA], LDA);
#pragma unroll
    for (int j = 0; j < 2; ++j) bF[j] = load_b_frag(&lB[(wn * 32 + j * 16) * LDB], LDB);
#pragma unroll
    for (int i = 0; i < 2; ++i)
#pragma unroll
      for (int j = 0; j < 2; ++j) acc[i][j] = wmma_bf16(aF[i], bF[j], acc[i][j]);
    __syncthreads();
  }

  const int half = lane >> 4, ln = lane & 15;
#pragma unroll
  for (int i = 0; i < 2; ++i)
#pragma unroll
    for (int j = 0; j < 2; ++j)
#pragma unroll
      for (int r = 0; r < 8; ++r) {
        const int m = mBase + wm * 32 + i * 16 + half * 8 + r;
        const int o = nBase + wn * 32 + j * 16 + ln;
        out[(size_t)m * CDIM + o] = acc[i][j][r] + bias[o];
      }
}

// ---------------------------------------------------------------------------
extern "C" void kernel_launch(void* const* d_in, const int* in_sizes, int n_in,
                              void* d_out, int out_size, void* d_ws, size_t ws_size,
                              hipStream_t stream) {
  const float* q  = (const float*)d_in[0];
  const float* k  = (const float*)d_in[1];
  const float* v  = (const float*)d_in[2];
  const float* Wq = (const float*)d_in[3];
  const float* Wk = (const float*)d_in[4];
  const float* Wv = (const float*)d_in[5];
  const float* Wo = (const float*)d_in[6];
  const float* bo = (const float*)d_in[7];
  float* out = (float*)d_out;

  const size_t per = (size_t)BH * SEQ * HDIM;
  bf16_t* q_ws  = (bf16_t*)d_ws;
  bf16_t* k_ws  = q_ws + per;
  bf16_t* vT_ws = k_ws + per;
  bf16_t* o_ws  = vT_ws + per;

  const dim3 gProj(BT / TILE_M, CDIM / TILE_N);
  proj_gemm_kernel<<<gProj, 256, 0, stream>>>(q, Wq, q_ws, 0);
  proj_gemm_kernel<<<gProj, 256, 0, stream>>>(k, Wk, k_ws, 0);
  proj_gemm_kernel<<<gProj, 256, 0, stream>>>(v, Wv, vT_ws, 2);

  attn_kernel<<<dim3(SEQ / 128, BH), 256, 0, stream>>>(q_ws, k_ws, vT_ws, o_ws);

  out_gemm_kernel<<<gProj, 256, 0, stream>>>(o_ws, Wo, bo, out);
}
